// InteractionEncoder_70566312673578
// MI455X (gfx1250) — compile-verified, matrix operand order checked by reading, not run on hardware
//
#include <hip/hip_runtime.h>
#include <hip/hip_bf16.h>
#include <cstdint>

// ---------------------------------------------------------------------------
// Types for CDNA5 WMMA (wave32): D(16x16 f32) = A(16x32 f16) x B(32x16 f16) + C
// ---------------------------------------------------------------------------
typedef __attribute__((ext_vector_type(16))) _Float16 v16h;
typedef __attribute__((ext_vector_type(8)))  _Float16 v8h;
typedef __attribute__((ext_vector_type(8)))  float    v8f;

union Frag16 { v16h v; v8h h[2]; };

static constexpr int B_  = 256;
static constexpr int L_  = 200;
static constexpr int D_  = 512;
static constexpr int BD  = B_ * D_;     // 131072
static constexpr int LD  = L_ * D_;     // 102400

__device__ __forceinline__ v8f wmma16(v16h a, v16h b, v8f c) {
  // v_wmma_f32_16x16x32_f16  (8-arg form: neg_a, A, neg_b, B, c_mod, C, reuse_a, reuse_b)
  return __builtin_amdgcn_wmma_f32_16x16x32_f16(false, a, false, b, (short)0, c, false, false);
}

// Fragment loader. Works for both A (row = m, ld = row stride) and B
// (row = weight-output-row n, contiguous along k) since the 16-bit A/B
// layouts are symmetric: lane r=lane&15 holds k in [k0+hi*8, +8) and
// [k0+16+hi*8, +8), hi = lane>>4.  All pointers are 16B aligned
// (ld multiple of 8 elements, k0 multiple of 32, base 256B aligned).
__device__ __forceinline__ v16h load_frag(const _Float16* __restrict__ base,
                                          int row0, int ld, int k0, int lane) {
  const int r  = lane & 15;
  const int hi = lane >> 4;
  const _Float16* p = base + (size_t)(row0 + r) * (size_t)ld + (size_t)(k0 + hi * 8);
  Frag16 f;
  f.h[0] = *(const v8h*)(p);
  f.h[1] = *(const v8h*)(p + 16);
  return f.v;
}

__device__ __forceinline__ float sigm(float x) { return 1.0f / (1.0f + __expf(-x)); }
__device__ __forceinline__ float tanh_fast(float x) {
  x = fminf(fmaxf(x, -15.0f), 15.0f);
  float e = __expf(2.0f * x);
  return (e - 1.0f) / (e + 1.0f);
}

// ---------------------------------------------------------------------------
// Utility kernels
// ---------------------------------------------------------------------------
__global__ void cvt_f32_to_f16(const float* __restrict__ in, _Float16* __restrict__ out, size_t n) {
  for (size_t i = (size_t)blockIdx.x * blockDim.x + threadIdx.x; i < n;
       i += (size_t)gridDim.x * blockDim.x)
    out[i] = (_Float16)in[i];
}

__global__ void fill_zero_u32(uint32_t* __restrict__ p, size_t n) {
  for (size_t i = (size_t)blockIdx.x * blockDim.x + threadIdx.x; i < n;
       i += (size_t)gridDim.x * blockDim.x)
    p[i] = 0u;
}

// ---------------------------------------------------------------------------
// GRU step (one launch per t).  Fuses the input projection (x_t @ wih^T) with
// the hidden projection (h @ whh^T):  grid = (B/64, D/16), block = (32,4).
// Each wave: 16 batch rows x 16 hidden cols, 4 accumulators (r, z, xn, hn).
// ---------------------------------------------------------------------------
__global__ void gru_step_kernel(const _Float16* __restrict__ hprev16,  // [B,512] f16
                                const _Float16* __restrict__ x16,      // session f16 + t*D, ld = L*D
                                const _Float16* __restrict__ whh16,    // [1536,512]
                                const _Float16* __restrict__ wih16,    // [1536,512]
                                const float*    __restrict__ bih,      // [1536]
                                const float*    __restrict__ bhh,      // [1536]
                                float*          __restrict__ h32,      // [B,512] in/out
                                _Float16*       __restrict__ out16) {  // s_seq slice t [B,512]
  const int lane = threadIdx.x;
  const int m0 = (blockIdx.x * blockDim.y + threadIdx.y) * 16;
  const int n0 = blockIdx.y * 16;

  v8f ar{}, az{}, axn{}, ahn{};
  const _Float16* wih_r = wih16;
  const _Float16* wih_z = wih16 + (size_t)512 * 512;
  const _Float16* wih_n = wih16 + (size_t)1024 * 512;
  const _Float16* whh_r = whh16;
  const _Float16* whh_z = whh16 + (size_t)512 * 512;
  const _Float16* whh_n = whh16 + (size_t)1024 * 512;

  for (int k0 = 0; k0 < 512; k0 += 32) {
    v16h a_h = load_frag(hprev16, m0, 512, k0, lane);
    v16h a_x = load_frag(x16,     m0, LD,  k0, lane);
    v16h b;
    b = load_frag(wih_r, n0, 512, k0, lane); ar  = wmma16(a_x, b, ar);
    b = load_frag(whh_r, n0, 512, k0, lane); ar  = wmma16(a_h, b, ar);
    b = load_frag(wih_z, n0, 512, k0, lane); az  = wmma16(a_x, b, az);
    b = load_frag(whh_z, n0, 512, k0, lane); az  = wmma16(a_h, b, az);
    b = load_frag(wih_n, n0, 512, k0, lane); axn = wmma16(a_x, b, axn);
    b = load_frag(whh_n, n0, 512, k0, lane); ahn = wmma16(a_h, b, ahn);
  }

  const int nc = n0 + (lane & 15);
  const int hi = lane >> 4;
  const float br  = bih[nc]        + bhh[nc];
  const float bz  = bih[512 + nc]  + bhh[512 + nc];
  const float bxn = bih[1024 + nc];
  const float bhn = bhh[1024 + nc];

#pragma unroll
  for (int r = 0; r < 8; ++r) {
    const int m = m0 + r + 8 * hi;
    const size_t idx = (size_t)m * 512 + nc;
    float rg = sigm(ar[r] + br);
    float zg = sigm(az[r] + bz);
    float ng = tanh_fast(axn[r] + bxn + rg * (ahn[r] + bhn));
    float hold = h32[idx];
    float hnew = (1.0f - zg) * ng + zg * hold;
    h32[idx]   = hnew;
    out16[idx] = (_Float16)hnew;
  }
}

// ---------------------------------------------------------------------------
// v[b,d] = sum_e w[d,e] * target[b,e]   (M=256, N=512, K=512)
// ---------------------------------------------------------------------------
__global__ void gemm_v_kernel(const _Float16* __restrict__ tgt16,  // [256,512]
                              const _Float16* __restrict__ w16,    // [512,512] row=d, contiguous e
                              float* __restrict__ v32) {           // [256,512]
  const int lane = threadIdx.x;
  const int m0 = (blockIdx.x * blockDim.y + threadIdx.y) * 16;
  const int n0 = blockIdx.y * 16;
  v8f c{};
  for (int k0 = 0; k0 < 512; k0 += 32) {
    v16h a = load_frag(tgt16, m0, 512, k0, lane);
    v16h b = load_frag(w16,   n0, 512, k0, lane);
    c = wmma16(a, b, c);
  }
  const int nc = n0 + (lane & 15);
  const int hi = lane >> 4;
#pragma unroll
  for (int r = 0; r < 8; ++r)
    v32[(size_t)(m0 + r + 8 * hi) * 512 + nc] = c[r];
}

// ---------------------------------------------------------------------------
// Attention logits + softmax over the sequence. One block per batch b.
// logits[l] = s_seq[l,b,:] . v[b,:]   then softmax over l.
// att stored time-major [L,B].
// ---------------------------------------------------------------------------
__global__ void attn_softmax_kernel(const _Float16* __restrict__ sseq16,  // [L,B,512]
                                    const float*    __restrict__ v32,    // [B,512]
                                    float*          __restrict__ att) {  // [L,B]
  __shared__ float logit[L_];
  const int b    = blockIdx.x;
  const int tid  = threadIdx.x;
  const int wave = tid >> 5;
  const int lane = tid & 31;
  const float* vb = v32 + (size_t)b * 512;

  for (int l = wave; l < L_; l += 8) {
    const _Float16* s = sseq16 + (size_t)l * BD + (size_t)b * 512;
    float acc = 0.0f;
    for (int d = lane; d < 512; d += 32) acc += (float)s[d] * vb[d];
#pragma unroll
    for (int off = 16; off > 0; off >>= 1) acc += __shfl_down(acc, off, 32);
    if (lane == 0) logit[l] = acc;
  }
  __syncthreads();

  if (tid < 32) {
    float mx = -1e30f;
    for (int l = tid; l < L_; l += 32) mx = fmaxf(mx, logit[l]);
#pragma unroll
    for (int off = 16; off > 0; off >>= 1) mx = fmaxf(mx, __shfl_xor(mx, off, 32));
    float sum = 0.0f;
    for (int l = tid; l < L_; l += 32) sum += __expf(logit[l] - mx);
#pragma unroll
    for (int off = 16; off > 0; off >>= 1) sum += __shfl_xor(sum, off, 32);
    const float inv = 1.0f / sum;
    for (int l = tid; l < L_; l += 32) att[(size_t)l * B_ + b] = __expf(logit[l] - mx) * inv;
  }
}

// ---------------------------------------------------------------------------
// AUGRU step, phase 1: r and u gates over ti = cat([h_prev, x_t]) (K=1024).
// Writes hr16 = (f16)(h * r) and au32 = att * u.
// ---------------------------------------------------------------------------
__global__ void augru_ru_kernel(const _Float16* __restrict__ hprev16,  // [B,512]
                                const _Float16* __restrict__ x16,      // s_seq slice t [B,512]
                                const _Float16* __restrict__ rw16,     // [512,1024]
                                const _Float16* __restrict__ uw16,     // [512,1024]
                                const float* __restrict__ rb,
                                const float* __restrict__ ub,
                                const float* __restrict__ h32,         // [B,512]
                                const float* __restrict__ att_t,       // [B]
                                _Float16* __restrict__ hr16,           // [B,512]
                                float*    __restrict__ au32) {         // [B,512]
  const int lane = threadIdx.x;
  const int m0 = (blockIdx.x * blockDim.y + threadIdx.y) * 16;
  const int n0 = blockIdx.y * 16;
  v8f cr{}, cu{};
  for (int k0 = 0; k0 < 1024; k0 += 32) {
    const _Float16* abase = (k0 < 512) ? hprev16 : x16;
    const int kk = (k0 < 512) ? k0 : (k0 - 512);
    v16h a = load_frag(abase, m0, 512, kk, lane);
    v16h b;
    b = load_frag(rw16, n0, 1024, k0, lane); cr = wmma16(a, b, cr);
    b = load_frag(uw16, n0, 1024, k0, lane); cu = wmma16(a, b, cu);
  }
  const int nc = n0 + (lane & 15);
  const int hi = lane >> 4;
  const float brb = rb[nc], bub = ub[nc];
#pragma unroll
  for (int r = 0; r < 8; ++r) {
    const int m = m0 + r + 8 * hi;
    const size_t idx = (size_t)m * 512 + nc;
    float rg = sigm(cr[r] + brb);
    float ug = sigm(cu[r] + bub);
    hr16[idx] = (_Float16)(h32[idx] * rg);
    au32[idx] = att_t[m] * ug;
  }
}

// ---------------------------------------------------------------------------
// AUGRU step, phase 2: hh = tanh(cat([h*r, x]) @ hhat_w^T + b); h update.
// ---------------------------------------------------------------------------
__global__ void augru_hh_kernel(const _Float16* __restrict__ hr16,   // [B,512]
                                const _Float16* __restrict__ x16,    // s_seq slice t
                                const _Float16* __restrict__ hw16,   // [512,1024]
                                const float* __restrict__ hb,
                                const float* __restrict__ au32,      // [B,512]
                                float*    __restrict__ h32,          // [B,512] in/out
                                _Float16* __restrict__ hnext16) {    // [B,512]
  const int lane = threadIdx.x;
  const int m0 = (blockIdx.x * blockDim.y + threadIdx.y) * 16;
  const int n0 = blockIdx.y * 16;
  v8f ch{};
  for (int k0 = 0; k0 < 1024; k0 += 32) {
    const _Float16* abase = (k0 < 512) ? hr16 : x16;
    const int kk = (k0 < 512) ? k0 : (k0 - 512);
    v16h a = load_frag(abase, m0, 512, kk, lane);
    v16h b = load_frag(hw16, n0, 1024, k0, lane);
    ch = wmma16(a, b, ch);
  }
  const int nc = n0 + (lane & 15);
  const int hi = lane >> 4;
  const float bb = hb[nc];
#pragma unroll
  for (int r = 0; r < 8; ++r) {
    const int m = m0 + r + 8 * hi;
    const size_t idx = (size_t)m * 512 + nc;
    float hh = tanh_fast(ch[r] + bb);
    float u  = au32[idx];
    float hnew = (1.0f - u) * h32[idx] + u * hh;
    h32[idx]     = hnew;
    hnext16[idx] = (_Float16)hnew;
  }
}

// ---------------------------------------------------------------------------
// Host orchestration
// ---------------------------------------------------------------------------
extern "C" void kernel_launch(void* const* d_in, const int* in_sizes, int n_in,
                              void* d_out, int out_size, void* d_ws, size_t ws_size,
                              hipStream_t stream) {
  (void)in_sizes; (void)n_in; (void)out_size; (void)ws_size;

  const float* session  = (const float*)d_in[0];   // [B,L,D]
  const float* target   = (const float*)d_in[1];   // [B,D]
  const float* w        = (const float*)d_in[2];   // [D,D]
  const float* gru_wih  = (const float*)d_in[3];   // [3D,D]
  const float* gru_whh  = (const float*)d_in[4];   // [3D,D]
  const float* gru_bih  = (const float*)d_in[5];   // [3D]
  const float* gru_bhh  = (const float*)d_in[6];   // [3D]
  const float* reset_w  = (const float*)d_in[7];   // [D,2D]
  const float* reset_b  = (const float*)d_in[8];
  const float* update_w = (const float*)d_in[9];
  const float* update_b = (const float*)d_in[10];
  const float* hhat_w   = (const float*)d_in[11];
  const float* hhat_b   = (const float*)d_in[12];

  // Bump allocator over d_ws (256B aligned chunks).
  uintptr_t cur = (uintptr_t)d_ws;
  auto alloc = [&](size_t bytes) -> void* {
    uintptr_t p = (cur + 255) & ~(uintptr_t)255;
    cur = p + bytes;
    return (void*)p;
  };

  _Float16* sess16 = (_Float16*)alloc((size_t)B_ * LD * 2);
  _Float16* wih16  = (_Float16*)alloc((size_t)1536 * 512 * 2);
  _Float16* whh16  = (_Float16*)alloc((size_t)1536 * 512 * 2);
  _Float16* w16    = (_Float16*)alloc((size_t)512 * 512 * 2);
  _Float16* tgt16  = (_Float16*)alloc((size_t)BD * 2);
  _Float16* rw16   = (_Float16*)alloc((size_t)512 * 1024 * 2);
  _Float16* uw16   = (_Float16*)alloc((size_t)512 * 1024 * 2);
  _Float16* hw16   = (_Float16*)alloc((size_t)512 * 1024 * 2);
  _Float16* sseq16 = (_Float16*)alloc((size_t)L_ * BD * 2);
  _Float16* h016   = (_Float16*)alloc((size_t)BD * 2);
  _Float16* hA16   = (_Float16*)alloc((size_t)BD * 2);
  _Float16* hB16   = (_Float16*)alloc((size_t)BD * 2);
  _Float16* hr16   = (_Float16*)alloc((size_t)BD * 2);
  float*    h32    = (float*)alloc((size_t)BD * 4);
  float*    v32    = (float*)alloc((size_t)BD * 4);
  float*    au32   = (float*)alloc((size_t)BD * 4);
  float*    att32  = (float*)alloc((size_t)L_ * B_ * 4);

  // --- convert everything to f16 once per launch ---
  auto cvt = [&](const float* src, _Float16* dst, size_t n) {
    unsigned grid = (unsigned)((n + 255) / 256);
    if (grid > 2048) grid = 2048;
    cvt_f32_to_f16<<<grid, 256, 0, stream>>>(src, dst, n);
  };
  cvt(session,  sess16, (size_t)B_ * LD);
  cvt(gru_wih,  wih16,  (size_t)1536 * 512);
  cvt(gru_whh,  whh16,  (size_t)1536 * 512);
  cvt(w,        w16,    (size_t)512 * 512);
  cvt(target,   tgt16,  (size_t)BD);
  cvt(reset_w,  rw16,   (size_t)512 * 1024);
  cvt(update_w, uw16,   (size_t)512 * 1024);
  cvt(hhat_w,   hw16,   (size_t)512 * 1024);

  // h0 = 0 (f32 running state and f16 copy for the first A-matrix)
  fill_zero_u32<<<256, 256, 0, stream>>>((uint32_t*)h32, (size_t)BD);
  fill_zero_u32<<<64, 256, 0, stream>>>((uint32_t*)h016, (size_t)BD / 2);

  const dim3 blk(32, 4);                 // 4 waves: 64 batch rows per block
  const dim3 grd(B_ / 64, D_ / 16);      // 4 x 32 blocks

  // --- GRU scan: 1 launch per step, input projection fused ---
  const _Float16* hp16 = h016;
  for (int t = 0; t < L_; ++t) {
    gru_step_kernel<<<grd, blk, 0, stream>>>(
        hp16, sess16 + (size_t)t * D_, whh16, wih16, gru_bih, gru_bhh,
        h32, sseq16 + (size_t)t * BD);
    hp16 = sseq16 + (size_t)t * BD;
  }

  // --- attention ---
  gemm_v_kernel<<<grd, blk, 0, stream>>>(tgt16, w16, v32);
  attn_softmax_kernel<<<B_, 256, 0, stream>>>(sseq16, v32, att32);

  // --- AUGRU scan: 2 launches per step (h*r feeds the hh GEMM) ---
  const _Float16* haug = sseq16 + (size_t)(L_ - 1) * BD;  // h_last (f16)
  for (int t = 0; t < L_; ++t) {
    _Float16* hnext = (t & 1) ? hB16 : hA16;
    const _Float16* xt = sseq16 + (size_t)t * BD;
    augru_ru_kernel<<<grd, blk, 0, stream>>>(
        haug, xt, rw16, uw16, reset_b, update_b, h32,
        att32 + (size_t)t * B_, hr16, au32);
    augru_hh_kernel<<<grd, blk, 0, stream>>>(
        hr16, xt, hw16, hhat_b, au32, h32, hnext);
    haug = hnext;
  }

  // --- output: final hidden state (fp32) ---
  hipMemcpyAsync(d_out, h32, (size_t)BD * sizeof(float),
                 hipMemcpyDeviceToDevice, stream);
}
